// Net2_52948356825336
// MI455X (gfx1250) — matrix operand order repaired and checked by reading, not compiled, not run on hardware
//
#include <hip/hip_runtime.h>

#define DEV __device__ __forceinline__

typedef __bf16 bf16;
typedef __attribute__((ext_vector_type(16))) __bf16 v16bf;
typedef __attribute__((ext_vector_type(8)))  __bf16 v8bf;
typedef __attribute__((ext_vector_type(4)))  __bf16 v4bf;
typedef __attribute__((ext_vector_type(2)))  __bf16 v2bf;
typedef __attribute__((ext_vector_type(8)))  float  v8f;
typedef __attribute__((ext_vector_type(4)))  int    v4i;

// Problem constants
constexpr int B_   = 32;
constexpr int S_   = 512;
constexpr int HS   = 640;
constexpr int NH   = 10;
constexpr int KD   = 64;
constexpr int MTOK = B_ * S_;   // 16384

// GEMM tiling
constexpr int BM = 128, BN = 128, BK = 32, LDT = 40;   // LDT padded for LDS banks
// Attention tiling
constexpr int QT = 128, KC = 64, LKV = 72;

DEV bf16 to_bf16_sw(float f) {
  union { float f; unsigned u; } x; x.f = f;
  unsigned r = x.u + 0x7fffu + ((x.u >> 16) & 1u);   // round-to-nearest-even
  unsigned short h = (unsigned short)(r >> 16);
  union { unsigned short s; bf16 b; } y; y.s = h;
  return y.b;
}

#if __has_builtin(__builtin_amdgcn_cvt_pk_bf16_f32)
DEV v2bf pk_bf16(float a, float b) {
  return __builtin_amdgcn_cvt_pk_bf16_f32(a, b);
}
#else
DEV v2bf pk_bf16(float a, float b) {
  v2bf r; r[0] = to_bf16_sw(a); r[1] = to_bf16_sw(b); return r;
}
#endif

DEV bf16 to_bf16(float f) { return pk_bf16(f, f)[0]; }

// ---------------------------------------------------------------------------
// Async HBM -> LDS staging (gfx1250 GLOBAL_LOAD_ASYNC_TO_LDS_B128, ASYNCcnt).
// Builtin signature (from clang diagnostic): (v4i AS1*, v4i AS3*, imm, imm).
// Falls back to a synchronous VGPR round-trip if the builtin is unavailable.
// ---------------------------------------------------------------------------
#if __has_builtin(__builtin_amdgcn_global_load_async_to_lds_b128)
#define HAVE_ASYNC_COPY 1
typedef __attribute__((address_space(1))) v4i gas_v4i;
typedef __attribute__((address_space(3))) v4i las_v4i;
DEV void async_b128(const bf16* g, bf16* l) {
  __builtin_amdgcn_global_load_async_to_lds_b128(
      (gas_v4i*)g, (las_v4i*)l, /*imm offset=*/0, /*cpol=*/0);
}
#else
#define HAVE_ASYNC_COPY 0
DEV void async_b128(const bf16* g, bf16* l) {
  *(v8bf*)l = *(const v8bf*)g;
}
#endif

DEV void async_wait_all() {
#if __has_builtin(__builtin_amdgcn_s_wait_asynccnt)
  __builtin_amdgcn_s_wait_asynccnt(0);
#elif HAVE_ASYNC_COPY
  asm volatile("s_wait_asynccnt 0x0" ::: "memory");
#endif
}

DEV v16bf combine16(v8bf lo, v8bf hi) {
  v16bf r;
#pragma unroll
  for (int i = 0; i < 8; ++i) { r[i] = lo[i]; r[i + 8] = hi[i]; }
  return r;
}

// A fragment (16x32 bf16): lane half selects K-block; row already folded into ptr.
// VGPR0..3 <- K = kh*8 + 0..7 ; VGPR4..7 <- K = 16 + kh*8 + 0..7
DEV v16bf load_frag_a(const bf16* row_ptr, int lane) {
  int kh = (lane >> 4) & 1;
  v8bf lo = *(const v8bf*)(row_ptr + kh * 8);
  v8bf hi = *(const v8bf*)(row_ptr + 16 + kh * 8);
  return combine16(lo, hi);
}

// B fragment (32x16 bf16), LDS stored n-major [N][LD] with K contiguous.
// VGPR0..3 <- K = kh*8 + 0..7 ; VGPR4..7 <- K = 16 + kh*8 + 0..7 ; column = lane&15 (in ptr).
DEV v16bf load_frag_b(const bf16* row_ptr, int lane) {
  int kh = (lane >> 4) & 1;
  v8bf lo = *(const v8bf*)(row_ptr + kh * 8);
  v8bf hi = *(const v8bf*)(row_ptr + 16 + kh * 8);
  return combine16(lo, hi);
}

DEV v8f wmma_bf16(v16bf a, v16bf b, v8f c) {
  return __builtin_amdgcn_wmma_f32_16x16x32_bf16(
      /*neg_a=*/false, a, /*neg_b=*/false, b,
      /*c_mod=*/(short)0, c, /*reuse_a=*/false, /*reuse_b=*/false);
}

DEV v8f vzero() { v8f z = {0.f, 0.f, 0.f, 0.f, 0.f, 0.f, 0.f, 0.f}; return z; }

// ---------------------------------------------------------------------------
// Kernel 0: transpose + convert weights to bf16, n-major [n_out][k_in].
// mats: 0=q_w, 1=k_w, 2=v_w (each [HS a][N*KD]) ; 3=o_w ([N*KD][HS o])
// ---------------------------------------------------------------------------
__global__ void prep_weights(const float* __restrict__ qw,
                             const float* __restrict__ kw,
                             const float* __restrict__ vw,
                             const float* __restrict__ ow,
                             bf16* __restrict__ wT) {
  int idx = blockIdx.x * blockDim.x + threadIdx.x;
  if (idx >= 4 * HS * HS) return;
  int mat = idx / (HS * HS);
  int rem = idx % (HS * HS);
  int r = rem / HS;   // input row  (k index of GEMM)
  int c = rem % HS;   // input col  (n index of GEMM)
  const float* src = (mat == 0) ? qw : (mat == 1) ? kw : (mat == 2) ? vw : ow;
  wT[(size_t)mat * HS * HS + (size_t)c * HS + r] = to_bf16(src[rem]);
}

// ---------------------------------------------------------------------------
// Kernel 1: fused QKV projection GEMM  [16384 x 640] * [640 x 1920] -> bf16
// output layout qkv[mat][b][h][s][kd] ; q pre-scaled by KD^-0.5
// ---------------------------------------------------------------------------
__global__ __launch_bounds__(256)
void qkv_gemm(const float* __restrict__ x, const bf16* __restrict__ wT,
              bf16* __restrict__ qkv) {
  __shared__ __align__(16) bf16 sA[BM * LDT];
  __shared__ __align__(16) bf16 sB[BN * LDT];
  const int tid  = threadIdx.x;
  const int lane = tid & 31;
  const int wave = tid >> 5;
  const int wm = wave >> 2;   // 0..1
  const int wn = wave & 3;    // 0..3
  const int m0 = blockIdx.y * BM;
  const int n0 = blockIdx.x * BN;

  v8f acc[4][2];
#pragma unroll
  for (int i = 0; i < 4; ++i)
#pragma unroll
    for (int j = 0; j < 2; ++j) acc[i][j] = vzero();

  for (int kt = 0; kt < HS; kt += BK) {
    __syncthreads();
    {   // stage B (bf16, n-major) by direct async HBM->LDS copies
      int seg  = (tid & 3) * 8;   // 8 bf16 = 16B segments of a 32-elem row
      int nrow = tid >> 2;        // 0..63
#pragma unroll
      for (int p = 0; p < 2; ++p) {
        int n = nrow + 64 * p;
        async_b128(wT + (size_t)(n0 + n) * HS + kt + seg, &sB[n * LDT + seg]);
      }
    }
    {   // stage A: f32 -> bf16, [BM][BK] (conversion must pass through VGPRs)
      int kp = tid & 15;     // k-pair
      int r0 = tid >> 4;     // 0..15
#pragma unroll
      for (int p = 0; p < 8; ++p) {
        int row = r0 + 16 * p;
        const float* g = x + (size_t)(m0 + row) * HS + kt + kp * 2;
        float2 f = *(const float2*)g;
        *(v2bf*)&sA[row * LDT + kp * 2] = pk_bf16(f.x, f.y);
      }
    }
    async_wait_all();
    __syncthreads();
    v16bf bfr[2];
#pragma unroll
    for (int j = 0; j < 2; ++j)
      bfr[j] = load_frag_b(&sB[(wn * 32 + j * 16 + (lane & 15)) * LDT], lane);
#pragma unroll
    for (int i = 0; i < 4; ++i) {
      v16bf afr = load_frag_a(&sA[(wm * 64 + i * 16 + (lane & 15)) * LDT], lane);
#pragma unroll
      for (int j = 0; j < 2; ++j) acc[i][j] = wmma_bf16(afr, bfr[j], acc[i][j]);
    }
  }
  // epilogue: scatter to [mat][b][h][s][kd]; convert in pairs (1 cvt / 2 elems)
#pragma unroll
  for (int i = 0; i < 4; ++i) {
#pragma unroll
    for (int j = 0; j < 2; ++j) {
      int ntot = n0 + wn * 32 + j * 16 + (lane & 15);
      int mat = ntot / HS;
      int rem = ntot % HS;
      int h = rem / KD, kd = rem % KD;
      float scale = (mat == 0) ? 0.125f : 1.0f;   // KD^-0.5
#pragma unroll
      for (int r = 0; r < 8; r += 2) {
        v2bf pr = pk_bf16(acc[i][j][r] * scale, acc[i][j][r + 1] * scale);
#pragma unroll
        for (int e = 0; e < 2; ++e) {
          int m = m0 + wm * 64 + i * 16 + r + e + ((lane >> 4) << 3);
          int b = m >> 9, s = m & (S_ - 1);
          size_t addr = ((((size_t)mat * B_ + b) * NH + h) * S_ + s) * KD + kd;
          qkv[addr] = pr[e];
        }
      }
    }
  }
}

// ---------------------------------------------------------------------------
// Kernel 2: flash attention. grid (S/QT, NH, B). 8 waves * 16 q-rows = 128.
// ---------------------------------------------------------------------------
__global__ __launch_bounds__(256)
void attn_kernel(const bf16* __restrict__ qkv,
                 const float* __restrict__ bias,   // [B,S]
                 const float* __restrict__ nb,     // [N,S,S]
                 bf16* __restrict__ wv)            // [B,S,HS]
{
  __shared__ __align__(16) bf16 kbuf[KC * LKV];
  __shared__ __align__(16) bf16 vbufT[KD * LKV];
  __shared__ __align__(16) bf16 pbuf[8][16 * LKV];

  const int tid  = threadIdx.x;
  const int lane = tid & 31;
  const int wave = tid >> 5;
  const int half = lane >> 4;
  const int qt = blockIdx.x, h = blockIdx.y, b = blockIdx.z;
  const int q0 = qt * QT;

  const bf16* gq = qkv + (((size_t)0 * B_ + b) * NH + h) * S_ * KD;
  const bf16* gk = qkv + (((size_t)1 * B_ + b) * NH + h) * S_ * KD;
  const bf16* gv = qkv + (((size_t)2 * B_ + b) * NH + h) * S_ * KD;

  // preload q A-fragments (16 rows per wave, K=64 -> two k-steps)
  const bf16* qrow = gq + (size_t)(q0 + wave * 16 + (lane & 15)) * KD;
  v16bf qa[2];
  qa[0] = load_frag_a(qrow, lane);
  qa[1] = load_frag_a(qrow + 32, lane);

  v8f acc[4];   // 4 kd-subtiles of 16
#pragma unroll
  for (int j = 0; j < 4; ++j) acc[j] = vzero();
  float mrow[8], lrow[8];
#pragma unroll
  for (int r = 0; r < 8; ++r) { mrow[r] = -3.0e38f; lrow[r] = 0.f; }

  for (int kc = 0; kc < S_ / KC; ++kc) {
    const int kb = kc * KC;
    __syncthreads();
    {   // stage K chunk row-major [key][kd] via async HBM->LDS (overlaps V work)
      int key = tid >> 2;
      int seg = (tid & 3) * 16;
      const bf16* g = gk + (size_t)(kb + key) * KD + seg;
      bf16* l = &kbuf[key * LKV + seg];
      async_b128(g, l);
      async_b128(g + 8, l + 8);
    }
    {   // stage V transposed [kd][key] (B-fragment wants n-major = kd rows)
      int key = tid >> 2;
      int kd0 = (tid & 3) * 16;
      const bf16* g = gv + (size_t)(kb + key) * KD + kd0;
      v8bf a = *(const v8bf*)g;
      v8bf c = *(const v8bf*)(g + 8);
#pragma unroll
      for (int e = 0; e < 8; ++e) {
        vbufT[(kd0 + e) * LKV + key]     = a[e];
        vbufT[(kd0 + 8 + e) * LKV + key] = c[e];
      }
    }
    async_wait_all();
    __syncthreads();

    // logits: 4 key-subtiles x 2 k-steps of WMMA
    v8f st[4];
#pragma unroll
    for (int j = 0; j < 4; ++j) {
      st[j] = vzero();
      const bf16* kp = &kbuf[(j * 16 + (lane & 15)) * LKV];
      v16bf b0 = load_frag_b(kp, lane);
      v16bf b1 = load_frag_b(kp + 32, lane);
      st[j] = wmma_bf16(qa[0], b0, st[j]);
      st[j] = wmma_bf16(qa[1], b1, st[j]);
    }
    // + bias (per key, broadcast) + nonbatched_bias[h][row][key]
#pragma unroll
    for (int j = 0; j < 4; ++j) {
      int col = kb + j * 16 + (lane & 15);
      float bv = bias[b * S_ + col];
      const float* nbp = nb + (size_t)h * S_ * S_ + col;
#pragma unroll
      for (int r = 0; r < 8; ++r) {
        int row = q0 + wave * 16 + r + (half << 3);
        st[j][r] += bv + nbp[(size_t)row * S_];
      }
    }
    // online softmax (row reductions across the 16-lane half)
#pragma unroll
    for (int r = 0; r < 8; ++r) {
      float rm = st[0][r];
#pragma unroll
      for (int j = 1; j < 4; ++j) rm = fmaxf(rm, st[j][r]);
      rm = fmaxf(rm, __shfl_xor(rm, 1, 32));
      rm = fmaxf(rm, __shfl_xor(rm, 2, 32));
      rm = fmaxf(rm, __shfl_xor(rm, 4, 32));
      rm = fmaxf(rm, __shfl_xor(rm, 8, 32));
      float mn = fmaxf(mrow[r], rm);
      float alpha = __expf(mrow[r] - mn);
      mrow[r] = mn;
      float rs = 0.f;
#pragma unroll
      for (int j = 0; j < 4; ++j) {
        float p = __expf(st[j][r] - mn);
        st[j][r] = p;
        rs += p;
      }
      rs += __shfl_xor(rs, 1, 32);
      rs += __shfl_xor(rs, 2, 32);
      rs += __shfl_xor(rs, 4, 32);
      rs += __shfl_xor(rs, 8, 32);
      lrow[r] = lrow[r] * alpha + rs;
#pragma unroll
      for (int j = 0; j < 4; ++j) acc[j][r] *= alpha;
    }
    // stage P (bf16) to per-wave LDS pad; pair conversions through cvt_pk
    bf16* pw = pbuf[wave];
#pragma unroll
    for (int j = 0; j < 4; ++j) {
      int col = j * 16 + (lane & 15);
#pragma unroll
      for (int r = 0; r < 8; r += 2) {
        v2bf pr = pk_bf16(st[j][r], st[j][r + 1]);
        int rl = r + (half << 3);
        pw[rl * LKV + col]       = pr[0];
        pw[(rl + 1) * LKV + col] = pr[1];
      }
    }
    // P . V : 2 k-steps x 4 kd-subtiles
#pragma unroll
    for (int ks = 0; ks < 2; ++ks) {
      v16bf pa = load_frag_a(&pw[(lane & 15) * LKV + ks * 32], lane);
#pragma unroll
      for (int j = 0; j < 4; ++j) {
        v16bf vb = load_frag_b(&vbufT[(j * 16 + (lane & 15)) * LKV + ks * 32], lane);
        acc[j] = wmma_bf16(pa, vb, acc[j]);
      }
    }
  }
  // finalize: divide by l, store bf16 wv[b][s][h*KD+kd]; paired conversions
#pragma unroll
  for (int r = 0; r < 8; r += 2) {
    float inv0 = 1.0f / lrow[r];
    float inv1 = 1.0f / lrow[r + 1];
    int row = q0 + wave * 16 + r + (half << 3);
#pragma unroll
    for (int j = 0; j < 4; ++j) {
      int kd = j * 16 + (lane & 15);
      v2bf pr = pk_bf16(acc[j][r] * inv0, acc[j][r + 1] * inv1);
      wv[((size_t)b * S_ + row) * HS + h * KD + kd]     = pr[0];
      wv[((size_t)b * S_ + row + 1) * HS + h * KD + kd] = pr[1];
    }
  }
}

// ---------------------------------------------------------------------------
// Kernel 3: output projection [16384 x 640] * [640 x 640] + o_b -> f32
// ---------------------------------------------------------------------------
__global__ __launch_bounds__(256)
void out_gemm(const bf16* __restrict__ wvm, const bf16* __restrict__ oT,
              const float* __restrict__ ob, float* __restrict__ out) {
  __shared__ __align__(16) bf16 sA[BM * LDT];
  __shared__ __align__(16) bf16 sB[BN * LDT];
  const int tid  = threadIdx.x;
  const int lane = tid & 31;
  const int wave = tid >> 5;
  const int wm = wave >> 2, wn = wave & 3;
  const int m0 = blockIdx.y * BM;
  const int n0 = blockIdx.x * BN;

  v8f acc[4][2];
#pragma unroll
  for (int i = 0; i < 4; ++i)
#pragma unroll
    for (int j = 0; j < 2; ++j) acc[i][j] = vzero();

  for (int kt = 0; kt < HS; kt += BK) {
    __syncthreads();
    {   // both tiles are bf16 in HBM: pure async HBM->LDS staging
      int seg = (tid & 3) * 8;   // 16B segments
      int row = tid >> 2;        // 0..63
#pragma unroll
      for (int p = 0; p < 2; ++p) {
        int r = row + 64 * p;
        async_b128(wvm + (size_t)(m0 + r) * HS + kt + seg, &sA[r * LDT + seg]);
        async_b128(oT  + (size_t)(n0 + r) * HS + kt + seg, &sB[r * LDT + seg]);
      }
    }
    async_wait_all();
    __syncthreads();
    v16bf bfr[2];
#pragma unroll
    for (int j = 0; j < 2; ++j)
      bfr[j] = load_frag_b(&sB[(wn * 32 + j * 16 + (lane & 15)) * LDT], lane);
#pragma unroll
    for (int i = 0; i < 4; ++i) {
      v16bf afr = load_frag_a(&sA[(wm * 64 + i * 16 + (lane & 15)) * LDT], lane);
#pragma unroll
      for (int j = 0; j < 2; ++j) acc[i][j] = wmma_bf16(afr, bfr[j], acc[i][j]);
    }
  }
#pragma unroll
  for (int i = 0; i < 4; ++i) {
#pragma unroll
    for (int j = 0; j < 2; ++j) {
      int n = n0 + wn * 32 + j * 16 + (lane & 15);
      float bv = ob[n];
#pragma unroll
      for (int r = 0; r < 8; ++r) {
        int m = m0 + wm * 64 + i * 16 + r + ((lane >> 4) << 3);
        out[(size_t)m * HS + n] = acc[i][j][r] + bv;
      }
    }
  }
}

// ---------------------------------------------------------------------------
extern "C" void kernel_launch(void* const* d_in, const int* in_sizes, int n_in,
                              void* d_out, int out_size, void* d_ws, size_t ws_size,
                              hipStream_t stream) {
  const float* q_data = (const float*)d_in[0];
  const float* bias   = (const float*)d_in[1];
  const float* nb     = (const float*)d_in[2];
  const float* qw     = (const float*)d_in[3];
  const float* kw     = (const float*)d_in[4];
  const float* vw     = (const float*)d_in[5];
  const float* ow     = (const float*)d_in[6];
  const float* ob     = (const float*)d_in[7];
  float* out = (float*)d_out;

  bf16* wT  = (bf16*)d_ws;                               // [4][640][640]
  bf16* qkv = wT + (size_t)4 * HS * HS;                  // [3][B][NH][S][KD]
  bf16* wvm = qkv + (size_t)3 * B_ * NH * S_ * KD;       // [B][S][HS]

  prep_weights<<<(4 * HS * HS + 255) / 256, 256, 0, stream>>>(qw, kw, vw, ow, wT);
  qkv_gemm<<<dim3(3 * HS / BN, MTOK / BM), 256, 0, stream>>>(q_data, wT, qkv);
  attn_kernel<<<dim3(S_ / QT, NH, B_), 256, 0, stream>>>(qkv, bias, nb, wvm);
  out_gemm<<<dim3(HS / BN, MTOK / BM), 256, 0, stream>>>(
      wvm, wT + (size_t)3 * HS * HS, ob, out);
}
